// GATLayer_80075370266888
// MI455X (gfx1250) — matrix operand order repaired and checked by reading, not compiled, not run on hardware
//
#include <hip/hip_runtime.h>

// Problem constants (from reference)
#define BT 8192
#define NN 17
#define CC 256
#define HH 4
#define DK 64
#define MROWS (BT * NN)   // 139264 rows = 1088 * 128
#define TILE_M 128
#define KSTEP 32
#define LDA 40            // LDS row stride in bf16 elements (80B = 20 banks, conflict-free)

typedef __attribute__((ext_vector_type(16))) __bf16 v16bf;
typedef __attribute__((ext_vector_type(8)))  float  v8f;

struct alignas(16) U4 { unsigned int a, b, c, d; };
struct alignas(16) F4 { float x, y, z, w; };
union Frag { U4 u[2]; v16bf v; };

// fp32 -> bf16 (round to nearest even), packed pair into one dword
__device__ __forceinline__ unsigned int pack_bf2(float lo, float hi) {
  union { float f; unsigned int u; } x, y;
  x.f = lo; y.f = hi;
  unsigned int ua = (x.u + 0x7FFFu + ((x.u >> 16) & 1u)) >> 16;
  unsigned int ub = (y.u + 0x7FFFu + ((y.u >> 16) & 1u)) >> 16;
  return (ua & 0xFFFFu) | (ub << 16);
}

// ---------------------------------------------------------------------------
// Kernel 1: h = x @ W^T  via v_wmma_f32_16x16x32_bf16, ping-pong LDS pipeline.
//   grid.x = MROWS/128 = 1088, block = 256 threads (8 waves)
//   wave w computes rows [w*16, w*16+16) x all 256 cols of its block tile.
//   Pipeline: global loads for tile k+1 issued BEFORE the 16 WMMAs of tile k;
//   convert + LDS store after; single barrier per K-step.
// ---------------------------------------------------------------------------
__global__ __launch_bounds__(256) void gat_gemm(const float* __restrict__ X,
                                                const float* __restrict__ W,
                                                float* __restrict__ Hout) {
  __shared__ unsigned short As[2][TILE_M * LDA];  // 2 x (128x32 bf16) of x
  __shared__ unsigned short Bs[2][CC * LDA];      // 2 x (256x32 bf16) of W rows

  const int tid  = threadIdx.x;
  const int lane = tid & 31;
  const int wv   = tid >> 5;
  const size_t m0 = (size_t)blockIdx.x * TILE_M;

  v8f acc[16] = {};

  const int ar = tid >> 1;          // A staging: row 0..127
  const int ah = (tid & 1) * 16;    // A staging: col 0 or 16
  const int br = tid;               // B staging: W row 0..255
  const float* sa = X + (m0 + ar) * CC + ah;   // +kk per step (16 floats)
  const float* sb = W + (size_t)br * CC;       // +kk per step (32 floats)

  // Fragment addressing (constant across K-steps)
  const int mr  = wv * 16 + (lane & 15);
  const int klo = (lane >> 4) << 3;      // A frag: K split 0..7 / 8..15 (+16..)
  const int kb  = (lane >> 4) << 3;      // B frag: K split 0..7 / 8..15 (+16..)

  // ---- prologue: stage K-tile 0 into buffer 0 ----
  {
    unsigned int pa[8], pb[16];
#pragma unroll
    for (int q = 0; q < 8; ++q)  pa[q] = pack_bf2(sa[2 * q], sa[2 * q + 1]);
#pragma unroll
    for (int q = 0; q < 16; ++q) pb[q] = pack_bf2(sb[2 * q], sb[2 * q + 1]);
    U4* da = (U4*)&As[0][ar * LDA + ah];
    U4 a0{pa[0], pa[1], pa[2], pa[3]}, a1{pa[4], pa[5], pa[6], pa[7]};
    da[0] = a0; da[1] = a1;
    U4* db = (U4*)&Bs[0][br * LDA];
#pragma unroll
    for (int q = 0; q < 4; ++q) {
      U4 t{pb[4 * q], pb[4 * q + 1], pb[4 * q + 2], pb[4 * q + 3]};
      db[q] = t;
    }
  }
  __syncthreads();

  int p = 0;
  for (int kk = 0; kk < CC; kk += KSTEP) {
    const bool more = (kk + KSTEP) < CC;

    // ---- issue global loads for NEXT tile (latency hidden behind WMMAs) ----
    float fa[16], fb[32];
    if (more) {
      const float* na = sa + kk + KSTEP;
      const float* nb = sb + kk + KSTEP;
#pragma unroll
      for (int q = 0; q < 4; ++q) {
        F4 t = *(const F4*)(na + 4 * q);
        fa[4 * q] = t.x; fa[4 * q + 1] = t.y; fa[4 * q + 2] = t.z; fa[4 * q + 3] = t.w;
      }
#pragma unroll
      for (int q = 0; q < 8; ++q) {
        F4 t = *(const F4*)(nb + 4 * q);
        fb[4 * q] = t.x; fb[4 * q + 1] = t.y; fb[4 * q + 2] = t.z; fb[4 * q + 3] = t.w;
      }
    }

    // ---- 16 WMMAs from buffer p ----
    Frag a;
    a.u[0] = *(const U4*)&As[p][mr * LDA + klo];
    a.u[1] = *(const U4*)&As[p][mr * LDA + klo + 16];
#pragma unroll
    for (int t = 0; t < 16; ++t) {
      const int nr = t * 16 + (lane & 15);
      Frag b;
      b.u[0] = *(const U4*)&Bs[p][nr * LDA + kb];
      b.u[1] = *(const U4*)&Bs[p][nr * LDA + kb + 16];
      acc[t] = __builtin_amdgcn_wmma_f32_16x16x32_bf16(
          false, a.v, false, b.v, (short)0, acc[t], false, false);
    }

    // ---- convert + store NEXT tile into buffer p^1 ----
    if (more) {
      unsigned int pa[8], pb[16];
#pragma unroll
      for (int q = 0; q < 8; ++q)  pa[q] = pack_bf2(fa[2 * q], fa[2 * q + 1]);
#pragma unroll
      for (int q = 0; q < 16; ++q) pb[q] = pack_bf2(fb[2 * q], fb[2 * q + 1]);
      U4* da = (U4*)&As[p ^ 1][ar * LDA + ah];
      U4 a0{pa[0], pa[1], pa[2], pa[3]}, a1{pa[4], pa[5], pa[6], pa[7]};
      da[0] = a0; da[1] = a1;
      U4* db = (U4*)&Bs[p ^ 1][br * LDA];
#pragma unroll
      for (int q = 0; q < 4; ++q) {
        U4 t{pb[4 * q], pb[4 * q + 1], pb[4 * q + 2], pb[4 * q + 3]};
        db[q] = t;
      }
    }
    __syncthreads();
    p ^= 1;
  }

  // ---- write D: VGPR v holds row v (lanes 0-15) / v+8 (lanes 16-31), col = lane&15
  const int ncol = lane & 15;
  const int mhi  = (lane >> 4) * 8;
  const size_t rbase = m0 + (size_t)wv * 16 + mhi;
#pragma unroll
  for (int t = 0; t < 16; ++t) {
#pragma unroll
    for (int v = 0; v < 8; ++v) {
      Hout[(rbase + v) * CC + t * 16 + ncol] = acc[t][v];
    }
  }
}

// ---------------------------------------------------------------------------
// Kernel 2: per-graph attention, in place on d_out (h lives there).
//   grid.x = BT, block = 128 threads. Each block owns one graph's 17x256 slice.
//   All bulk global traffic uses 16B vector accesses.
// ---------------------------------------------------------------------------
__global__ __launch_bounds__(128) void gat_attn(const float* __restrict__ Hin,
                                                const float* __restrict__ X,
                                                const int* __restrict__ Adj,
                                                const float* __restrict__ Aw,
                                                float* __restrict__ Out) {
  __shared__ __align__(16) float hbuf[NN * CC];   // 17x256
  __shared__ float ebuf[NN * NN * HH];            // [i][j*4+h]
  __shared__ float eib[NN * HH];
  __shared__ float ejb[NN * HH];
  __shared__ float sbuf[NN * HH];
  __shared__ float awb[2 * DK];

  const int b   = blockIdx.x;
  const int tid = threadIdx.x;
  const size_t base = (size_t)b * (NN * CC);
  const int NV4 = (NN * CC) / 4;                  // 1088 float4 per graph

  {
    const F4* hg = (const F4*)(Hin + base);
    F4* hs = (F4*)hbuf;
    for (int i = tid; i < NV4; i += 128) hs[i] = hg[i];
  }
  if (tid < 2 * DK) awb[tid] = Aw[tid];
  __syncthreads();

  // ei[n,h] = sum_d h[n,h,d]*a1[d];  ej[n,h] = sum_d h[n,h,d]*a2[d]
  if (tid < NN * HH) {
    const int n = tid >> 2, h = tid & 3;
    const float* hp = &hbuf[n * CC + h * DK];
    float s1 = 0.f, s2 = 0.f;
#pragma unroll 8
    for (int d = 0; d < DK; ++d) { s1 += hp[d] * awb[d]; s2 += hp[d] * awb[DK + d]; }
    eib[tid] = s1; ejb[tid] = s2;
  }
  __syncthreads();

  // e[i,j,h] = leakyrelu(ei+ej) + mask
  for (int idx = tid; idx < NN * NN * HH; idx += 128) {
    const int i = idx / (NN * HH);
    const int r = idx - i * (NN * HH);
    const int j = r >> 2, h = r & 3;
    float e = eib[i * HH + h] + ejb[j * HH + h];
    e = (e >= 0.f) ? e : 0.2f * e;
    if (Adj[(size_t)b * (NN * NN) + i * NN + j] == 0) e += -1.0e9f;
    ebuf[idx] = e;
  }
  __syncthreads();

  // softmax over j per (i,h); each thread owns one (i,h) row exclusively
  if (tid < NN * HH) {
    const int i = tid >> 2, h = tid & 3;
    float m = -3.0e38f;
    for (int j = 0; j < NN; ++j) m = fmaxf(m, ebuf[i * NN * HH + j * HH + h]);
    float s = 0.f;
    for (int j = 0; j < NN; ++j) {
      float t = __expf(ebuf[i * NN * HH + j * HH + h] - m);
      ebuf[i * NN * HH + j * HH + h] = t;
      s += t;
    }
    const float inv = 1.f / s;
    for (int j = 0; j < NN; ++j) ebuf[i * NN * HH + j * HH + h] *= inv;
  }
  __syncthreads();

  // s[j,h] = sum_i alpha[i,j,h]
  if (tid < NN * HH) {
    const int j = tid >> 2, h = tid & 3;
    float s = 0.f;
    for (int i = 0; i < NN; ++i) s += ebuf[i * NN * HH + j * HH + h];
    sbuf[tid] = s;
  }
  __syncthreads();

  // out = h * s + x   (vectorized: a float4 never crosses a 64-elem head block)
  {
    const F4* xg = (const F4*)(X + base);
    F4* og = (F4*)(Out + base);
    const F4* hs = (const F4*)hbuf;
    for (int i = tid; i < NV4; i += 128) {
      const int n = i >> 6;           // (i*4) / 256
      const int h = (i >> 4) & 3;     // ((i*4) / 64) % 4
      const float s = sbuf[n * HH + h];
      F4 h4 = hs[i];
      F4 x4 = xg[i];
      F4 o{h4.x * s + x4.x, h4.y * s + x4.y, h4.z * s + x4.z, h4.w * s + x4.w};
      og[i] = o;
    }
  }
}

// ---------------------------------------------------------------------------
extern "C" void kernel_launch(void* const* d_in, const int* in_sizes, int n_in,
                              void* d_out, int out_size, void* d_ws, size_t ws_size,
                              hipStream_t stream) {
  (void)in_sizes; (void)n_in; (void)out_size; (void)d_ws; (void)ws_size;
  const float* x   = (const float*)d_in[0];   // (8192,17,256) f32
  const int*   adj = (const int*)d_in[1];     // (8192,17,17) i32
  const float* Ww  = (const float*)d_in[2];   // (256,256) f32
  const float* aw  = (const float*)d_in[3];   // (1,128) f32
  float* out = (float*)d_out;                 // (8192,17,256) f32

  // h = x @ W^T  written into d_out, then attention rewrites d_out in place.
  gat_gemm<<<dim3(MROWS / TILE_M), dim3(256), 0, stream>>>(x, Ww, out);
  gat_attn<<<dim3(BT), dim3(128), 0, stream>>>(out, x, adj, aw, out);
}
